// AttentionLayers_82171314307467
// MI455X (gfx1250) — matrix operand order repaired and checked by reading, not compiled, not run on hardware
//
#include <hip/hip_runtime.h>

// ---------------------------------------------------------------------------
// Bahdanau attention on MI455X (gfx1250, wave32, WMMA).
// B=64, L=1024, H=512, E=2H=1024.
//
// att  = relu(enc @ W2 + (hidden @ W1 + b)) @ Wv   (bf16 WMMA, f32 acc, fused)
// out  = softmax(att) @ enc                        (bandwidth bound)
// ---------------------------------------------------------------------------

#define B_   64
#define L_   1024
#define H_   512
#define E_   1024    // 2H
#define K2_  1024    // GEMM reduction dim (enc @ W_attn[H:3H])

typedef __attribute__((ext_vector_type(16))) __bf16          v16bf;
typedef __attribute__((ext_vector_type(16))) unsigned short  v16u;
typedef __attribute__((ext_vector_type(8)))  float           v8f;

// float -> bf16 (round-to-nearest-even), raw bits
__device__ __forceinline__ unsigned short f2bf_raw(float f) {
    unsigned u = __builtin_bit_cast(unsigned, f);
    u += 0x7FFFu + ((u >> 16) & 1u);
    return (unsigned short)(u >> 16);
}

// pack 16 f32 (A-fragment per 16-bit 16x32 WMMA layout) into v16bf
__device__ __forceinline__ v16bf cvt_a(float4 x0, float4 x1, float4 x2, float4 x3) {
    v16u au;
    au[0]  = f2bf_raw(x0.x); au[1]  = f2bf_raw(x0.y);
    au[2]  = f2bf_raw(x0.z); au[3]  = f2bf_raw(x0.w);
    au[4]  = f2bf_raw(x1.x); au[5]  = f2bf_raw(x1.y);
    au[6]  = f2bf_raw(x1.z); au[7]  = f2bf_raw(x1.w);
    au[8]  = f2bf_raw(x2.x); au[9]  = f2bf_raw(x2.y);
    au[10] = f2bf_raw(x2.z); au[11] = f2bf_raw(x2.w);
    au[12] = f2bf_raw(x3.x); au[13] = f2bf_raw(x3.y);
    au[14] = f2bf_raw(x3.z); au[15] = f2bf_raw(x3.w);
    return __builtin_bit_cast(v16bf, au);
}

// ---------------------------------------------------------------------------
// Kernel 1a: W2T[n][k] = bf16( W_attn[H + k][n] ) — transposed bf16 weights,
// so each lane's B-fragment is one contiguous 32B run. One-off, 1 MB.
// ---------------------------------------------------------------------------
__global__ void __launch_bounds__(256)
prep_w2t(const float* __restrict__ W_attn, unsigned short* __restrict__ w2t) {
    int idx = blockIdx.x * 256 + threadIdx.x;      // 512 * 1024 total
    int n = idx >> 10;
    int k = idx & (K2_ - 1);
    w2t[idx] = f2bf_raw(W_attn[(size_t)(H_ + k) * H_ + n]);
}

// ---------------------------------------------------------------------------
// Kernel 1b: hb[b][n] = hidden[b] @ W_attn[0:H] + b_attn  (L-invariant term)
// ---------------------------------------------------------------------------
__global__ void __launch_bounds__(512)
prep_hb(const float* __restrict__ hidden, const float* __restrict__ W_attn,
        const float* __restrict__ b_attn, float* __restrict__ hb) {
    int b = blockIdx.x;
    int n = threadIdx.x;                            // H_ = 512 threads
    float s = b_attn[n];
    for (int k = 0; k < H_; ++k)
        s = fmaf(hidden[b * H_ + k], W_attn[(size_t)k * H_ + n], s);
    hb[b * H_ + n] = s;
}

// ---------------------------------------------------------------------------
// Kernel 2: fused  att[b][l] = relu(enc[b,l] @ W2 + hb[b]) @ W_v
// Grid (L/16, B), 256 threads = 8 waves; wave w owns N in [w*64, w*64+64).
// Software-pipelined: loads for K-step k+1 issue before WMMAs of step k.
// ---------------------------------------------------------------------------
__global__ void __launch_bounds__(256)
attn_energy(const float* __restrict__ enc, const unsigned short* __restrict__ w2t,
            const float* __restrict__ hb, const float* __restrict__ Wv,
            float* __restrict__ att) {
    const int b    = blockIdx.y;
    const int m0   = blockIdx.x * 16;
    const int lane = threadIdx.x & 31;
    const int wave = threadIdx.x >> 5;

    __shared__ float lds_att[16];
    if (threadIdx.x < 16) lds_att[threadIdx.x] = 0.0f;
    __syncthreads();

    const int row  = lane & 15;     // A: M row / B: column-in-tile / D: N lane
    const int half = lane >> 4;     // K-half selector per ISA VGPR layouts
    const int n0   = wave * 64;

    // A source: row (b, m0+row); lane covers K = kk+half*16+{0..15}
    const float* arow = enc + ((size_t)b * L_ + (size_t)(m0 + row)) * E_ + half * 16;
    // B source: column n0+row of W2T; K = kk + half*16 + {0..15} contiguous
    const unsigned short* bb = w2t + (size_t)(n0 + row) * K2_ + half * 16;

    v8f acc0 = {}, acc1 = {}, acc2 = {}, acc3 = {};

    // ---- prologue: loads for K-step 0 ----
    const float4* pa = (const float4*)(arow);
    float4 x0 = pa[0], x1 = pa[1], x2 = pa[2], x3 = pa[3];
    v16bf nb0 = *(const v16bf*)(bb);
    v16bf nb1 = *(const v16bf*)(bb + 16 * K2_);
    v16bf nb2 = *(const v16bf*)(bb + 32 * K2_);
    v16bf nb3 = *(const v16bf*)(bb + 48 * K2_);

    // ---- pipelined main loop: kn is the K-step being LOADED ----
    for (int kn = 32; kn < K2_; kn += 32) {
        v16bf a  = cvt_a(x0, x1, x2, x3);
        v16bf b0 = nb0, b1 = nb1, b2 = nb2, b3 = nb3;

        __builtin_prefetch((const void*)(arow + kn + 64), 0, 1);  // speculative

        const float4* pn = (const float4*)(arow + kn);
        x0 = pn[0]; x1 = pn[1]; x2 = pn[2]; x3 = pn[3];
        const unsigned short* bn = bb + kn;
        nb0 = *(const v16bf*)(bn);
        nb1 = *(const v16bf*)(bn + 16 * K2_);
        nb2 = *(const v16bf*)(bn + 32 * K2_);
        nb3 = *(const v16bf*)(bn + 48 * K2_);

        acc0 = __builtin_amdgcn_wmma_f32_16x16x32_bf16(false, a, false, b0, (short)0, acc0, false, false);
        acc1 = __builtin_amdgcn_wmma_f32_16x16x32_bf16(false, a, false, b1, (short)0, acc1, false, false);
        acc2 = __builtin_amdgcn_wmma_f32_16x16x32_bf16(false, a, false, b2, (short)0, acc2, false, false);
        acc3 = __builtin_amdgcn_wmma_f32_16x16x32_bf16(false, a, false, b3, (short)0, acc3, false, false);
    }
    // ---- epilogue K-step (data loaded for kk = K2_-32) ----
    {
        v16bf a = cvt_a(x0, x1, x2, x3);
        acc0 = __builtin_amdgcn_wmma_f32_16x16x32_bf16(false, a, false, nb0, (short)0, acc0, false, false);
        acc1 = __builtin_amdgcn_wmma_f32_16x16x32_bf16(false, a, false, nb1, (short)0, acc1, false, false);
        acc2 = __builtin_amdgcn_wmma_f32_16x16x32_bf16(false, a, false, nb2, (short)0, acc2, false, false);
        acc3 = __builtin_amdgcn_wmma_f32_16x16x32_bf16(false, a, false, nb3, (short)0, acc3, false, false);
    }

    // ---- fused epilogue: relu(acc + hb) . Wv, reduced over N ---------------
    // D layout: lane<16 -> N=lane, VGPR r -> M=r ; lane>=16 -> N=lane-16, M=r+8
    float rowsum[8];
#pragma unroll
    for (int r = 0; r < 8; ++r) rowsum[r] = 0.0f;

#define EPI(ACC, T)                                                        \
    {                                                                      \
        const int n  = n0 + (T) * 16 + row;                                \
        const float hbn = hb[b * H_ + n];                                  \
        const float wv  = Wv[n];                                           \
        _Pragma("unroll")                                                  \
        for (int r = 0; r < 8; ++r) {                                      \
            float e = ACC[r] + hbn;                                        \
            e = e > 0.0f ? e : 0.0f;                                       \
            rowsum[r] = fmaf(e, wv, rowsum[r]);                            \
        }                                                                  \
    }
    EPI(acc0, 0) EPI(acc1, 1) EPI(acc2, 2) EPI(acc3, 3)
#undef EPI

    // butterfly reduce over the 16 N-lanes of each half (stays within half)
#pragma unroll
    for (int r = 0; r < 8; ++r) {
        float v = rowsum[r];
        v += __shfl_xor(v, 1);
        v += __shfl_xor(v, 2);
        v += __shfl_xor(v, 4);
        v += __shfl_xor(v, 8);
        rowsum[r] = v;
    }
    if (row == 0) {             // lanes 0 (rows 0..7) and 16 (rows 8..15)
#pragma unroll
        for (int r = 0; r < 8; ++r)
            atomicAdd(&lds_att[half * 8 + r], rowsum[r]);   // ds_add_f32
    }
    __syncthreads();
    if (threadIdx.x < 16)
        att[(size_t)b * L_ + m0 + threadIdx.x] = lds_att[threadIdx.x];
}

// ---------------------------------------------------------------------------
// Kernel 3: softmax over L per batch (exp stored, then scaled in place).
// ---------------------------------------------------------------------------
__global__ void __launch_bounds__(256)
softmax_k(const float* __restrict__ att, float* __restrict__ wts) {
    const int b = blockIdx.x, tid = threadIdx.x;
    __shared__ float red[256];
    const float* arow = att + (size_t)b * L_;
    float* wrow = wts + (size_t)b * L_;

    float m = -3.402823466e38f;
    for (int i = tid; i < L_; i += 256) m = fmaxf(m, arow[i]);
    red[tid] = m; __syncthreads();
    for (int s = 128; s > 0; s >>= 1) {
        if (tid < s) red[tid] = fmaxf(red[tid], red[tid + s]);
        __syncthreads();
    }
    m = red[0]; __syncthreads();

    float sum = 0.0f;
    for (int i = tid; i < L_; i += 256) {
        float e = __expf(arow[i] - m);
        wrow[i] = e;
        sum += e;
    }
    red[tid] = sum; __syncthreads();
    for (int s = 128; s > 0; s >>= 1) {
        if (tid < s) red[tid] += red[tid + s];
        __syncthreads();
    }
    const float inv = 1.0f / red[0];
    for (int i = tid; i < L_; i += 256) wrow[i] *= inv;
}

// ---------------------------------------------------------------------------
// Kernel 4: context[b][e] = sum_l w[b][l] * enc[b][l][e]  (bandwidth bound)
// Grid (E/256, B). Weights row staged in LDS.
// ---------------------------------------------------------------------------
__global__ void __launch_bounds__(256)
context_k(const float* __restrict__ enc, const float* __restrict__ wts,
          float* __restrict__ out) {
    const int b = blockIdx.y;
    const int e = blockIdx.x * 256 + threadIdx.x;
    __shared__ float w[L_];
    for (int i = threadIdx.x; i < L_; i += 256) w[i] = wts[(size_t)b * L_ + i];
    __syncthreads();

    const float* ebase = enc + (size_t)b * L_ * E_ + e;
    float acc = 0.0f;
#pragma unroll 4
    for (int l = 0; l < L_; ++l)
        acc = fmaf(w[l], ebase[(size_t)l * E_], acc);
    out[(size_t)b * E_ + e] = acc;
}

// ---------------------------------------------------------------------------
// Launch. Workspace layout (bytes):
//   [0, 1MB)        : W2T bf16   (512*1024 u16)
//   [1MB, +128KB)   : hb  f32    (64*512)
//   [.., +256KB)    : att f32    (64*1024)
//   [.., +256KB)    : wts f32    (64*1024)
// ---------------------------------------------------------------------------
extern "C" void kernel_launch(void* const* d_in, const int* in_sizes, int n_in,
                              void* d_out, int out_size, void* d_ws, size_t ws_size,
                              hipStream_t stream) {
    const float* hidden = (const float*)d_in[0];
    const float* enc    = (const float*)d_in[1];
    const float* W_attn = (const float*)d_in[2];
    const float* b_attn = (const float*)d_in[3];
    const float* W_v    = (const float*)d_in[4];
    float* out = (float*)d_out;

    char* ws = (char*)d_ws;
    unsigned short* w2t = (unsigned short*)ws;
    float* hb  = (float*)(ws + (1u << 20));
    float* att = (float*)(ws + (1u << 20) + (128u << 10));
    float* wts = (float*)(ws + (1u << 20) + (128u << 10) + (256u << 10));

    hipLaunchKernelGGL(prep_w2t,    dim3((H_ * K2_) / 256), dim3(256), 0, stream, W_attn, w2t);
    hipLaunchKernelGGL(prep_hb,     dim3(B_),               dim3(512), 0, stream, hidden, W_attn, b_attn, hb);
    hipLaunchKernelGGL(attn_energy, dim3(L_ / 16, B_),      dim3(256), 0, stream, enc, w2t, hb, W_v, att);
    hipLaunchKernelGGL(softmax_k,   dim3(B_),               dim3(256), 0, stream, att, wts);
    hipLaunchKernelGGL(context_k,   dim3(E_ / 256, B_),     dim3(256), 0, stream, enc, wts, out);
}
